// TGCRNCell_29411936043051
// MI455X (gfx1250) — matrix operand order repaired and checked by reading, not compiled, not run on hardware
//
#include <hip/hip_runtime.h>
#include <hip/hip_bf16.h>

// Problem constants (from reference setup_inputs)
#define BDIM 32
#define NN   2048
#define DD   64
#define EE   10

typedef __attribute__((ext_vector_type(16))) __bf16 v16bf;
typedef __attribute__((ext_vector_type(8)))  __bf16 v8bf;
typedef __attribute__((ext_vector_type(8)))  float  v8f;

union ABfrag { v16bf v; v8bf h[2]; };

__device__ __forceinline__ __bf16 f2bf(float f) { return (__bf16)f; }
__device__ __forceinline__ float  bf2f(__bf16 b) { return (float)b; }

// ---------------------------------------------------------------------------
// CDNA5 async global->LDS copy (16 bytes per lane), tracked by ASYNCcnt.
// GV addressing mode: per-lane 64-bit global address, LDS dest offset in VGPR.
// ---------------------------------------------------------------------------
__device__ __forceinline__ void async_ld_b128(void* lds, const void* g) {
  unsigned l = (unsigned)(uintptr_t)lds;   // low 32 bits = LDS byte offset
  asm volatile("global_load_async_to_lds_b128 %0, %1, off"
               :: "v"(l), "v"(g) : "memory");
}
__device__ __forceinline__ void wait_async0() {
  asm volatile("s_wait_asynccnt 0" ::: "memory");
}

// ---------------------------------------------------------------------------
// adj = softmax(relu(emb @ emb^T)) row-wise, stored bf16 row-major.
// One block (256 threads) per row; 8 columns per thread held in registers.
// ---------------------------------------------------------------------------
__global__ void adj_softmax_kernel(const float* __restrict__ emb,
                                   __bf16* __restrict__ adj) {
  const int n   = blockIdx.x;
  const int tid = threadIdx.x;
  __shared__ float en[EE];
  __shared__ float red[256];
  if (tid < EE) en[tid] = emb[n * EE + tid];
  __syncthreads();

  float v[8];
  float lmax = -1e30f;
#pragma unroll
  for (int i = 0; i < 8; ++i) {
    const int m = tid + i * 256;
    const float* em = emb + (size_t)m * EE;
    float d = 0.f;
#pragma unroll
    for (int e = 0; e < EE; ++e) d += en[e] * em[e];
    d = fmaxf(d, 0.f);      // relu
    v[i] = d;
    lmax = fmaxf(lmax, d);
  }
  red[tid] = lmax; __syncthreads();
  for (int s = 128; s > 0; s >>= 1) {
    if (tid < s) red[tid] = fmaxf(red[tid], red[tid + s]);
    __syncthreads();
  }
  const float bmax = red[0]; __syncthreads();

  float lsum = 0.f;
#pragma unroll
  for (int i = 0; i < 8; ++i) { v[i] = __expf(v[i] - bmax); lsum += v[i]; }
  red[tid] = lsum; __syncthreads();
  for (int s = 128; s > 0; s >>= 1) {
    if (tid < s) red[tid] += red[tid + s];
    __syncthreads();
  }
  const float inv = 1.f / red[0];
#pragma unroll
  for (int i = 0; i < 8; ++i)
    adj[(size_t)n * NN + tid + i * 256] = f2bf(v[i] * inv);
}

// ---------------------------------------------------------------------------
// x (f32) -> x_bf16 ; also seed candidate[..., 0:64] = x
// ---------------------------------------------------------------------------
__global__ void cvt_x_kernel(const float* __restrict__ x,
                             __bf16* __restrict__ xbf,
                             __bf16* __restrict__ cand) {
  const int i = blockIdx.x * blockDim.x + threadIdx.x;
  if (i >= BDIM * NN * DD) return;
  const __bf16 b = f2bf(x[i]);
  xbf[i] = b;
  const int d  = i & 63;
  const int bn = i >> 6;
  cand[(size_t)bn * 128 + d] = b;
}

__global__ void cvt_f2bf_kernel(const float* __restrict__ src,
                                __bf16* __restrict__ dst, int nelem) {
  const int i = blockIdx.x * blockDim.x + threadIdx.x;
  if (i < nelem) dst[i] = f2bf(src[i]);
}

// bias[n][o] = sum_e emb[n][e] * pool[e][o]
__global__ void bias_kernel(const float* __restrict__ emb,
                            const float* __restrict__ pool,
                            float* __restrict__ bias, int dout) {
  const int i = blockIdx.x * blockDim.x + threadIdx.x;
  if (i >= NN * dout) return;
  const int n = i / dout, o = i - n * dout;
  float s = 0.f;
#pragma unroll
  for (int e = 0; e < EE; ++e) s += emb[n * EE + e] * pool[e * dout + o];
  bias[i] = s;
}

// ---------------------------------------------------------------------------
// Y[b] = alpha * (adj @ X[b]) + beta * ref[b]    (all bf16, f32 accumulate)
// Block: 256 threads = 8 waves, covers 32 nodes x W columns.
//   W=64 : wave -> (node-tile = wid/4, col-tile = wid%4)
//   W=128: wave -> col-tile = wid, loops both node-tiles (2 accumulators)
// A chunk (32 nodes x 32 K) staged row-major in LDS via ASYNC global->LDS.
// X chunk (32 x W) staged in LDS, K-transposed for contiguous B-fragments.
// ---------------------------------------------------------------------------
template <int W>
__global__ void spmm_kernel(const __bf16* __restrict__ adj,
                            const __bf16* __restrict__ X,
                            float alpha, float beta,
                            const __bf16* __restrict__ ref,
                            __bf16* __restrict__ Y) {
  constexpr int NTA = (W == 64) ? 1 : 2;
  __shared__ __bf16 ldsA[32 * 32];    // ldsA[m*32 + k]   (row-major, async)
  __shared__ __bf16 ldsB[32 * W];     // ldsB[j*32 + kk]  (K-transposed)
  const int tid  = threadIdx.x;
  const int lane = tid & 31, wid = tid >> 5;
  const int b    = blockIdx.y;
  const int n0   = blockIdx.x * 32;
  const int j0   = ((W == 64) ? (wid & 3) : wid) * 16;
  const int jl    = lane & 15;
  const int abase = (lane < 16) ? 0 : 8;
  const int bbase = (lane < 16) ? 0 : 8;
  const __bf16* Xb = X + (size_t)b * NN * W;

  v8f acc[NTA];
#pragma unroll
  for (int t = 0; t < NTA; ++t)
#pragma unroll
    for (int r = 0; r < 8; ++r) acc[t][r] = 0.f;

  for (int K0 = 0; K0 < NN; K0 += 32) {
    // --- async stage of A (adj rows): 32 rows x 64 bytes = 128 b128 units ---
    if (tid < 128) {
      const int m    = tid >> 2;
      const int part = (tid & 3) * 8;
      async_ld_b128(&ldsA[m * 32 + part],
                    adj + (size_t)(n0 + m) * NN + K0 + part);
    }
    // --- manual transpose stage of B (X chunk) ---
    for (int idx = tid; idx < 32 * W; idx += blockDim.x) {
      const int kk = idx / W, j = idx - kk * W;
      ldsB[j * 32 + kk] = Xb[(size_t)(K0 + kk) * W + j];
    }
    if (K0 + 32 < NN)
      __builtin_prefetch(&Xb[(size_t)(K0 + 32) * W + tid], 0, 1);
    wait_async0();
    __syncthreads();

    ABfrag bf_;
    const __bf16* bp = &ldsB[(j0 + jl) * 32 + bbase];
    bf_.h[0] = *(const v8bf*)bp;
    bf_.h[1] = *(const v8bf*)(bp + 16);

#pragma unroll
    for (int t = 0; t < NTA; ++t) {
      const int nt = (W == 64) ? (wid >> 2) : t;
      const __bf16* ap = &ldsA[(nt * 16 + jl) * 32 + abase];
      ABfrag af;
      af.h[0] = *(const v8bf*)ap;
      af.h[1] = *(const v8bf*)(ap + 16);
      acc[t] = __builtin_amdgcn_wmma_f32_16x16x32_bf16(
          false, af.v, false, bf_.v, (short)0, acc[t], false, false);
    }
    __syncthreads();
  }

#pragma unroll
  for (int t = 0; t < NTA; ++t) {
    const int nt = (W == 64) ? (wid >> 2) : t;
#pragma unroll
    for (int r = 0; r < 8; ++r) {
      const int m = n0 + nt * 16 + r + ((lane >= 16) ? 8 : 0);
      const size_t o = ((size_t)b * NN + m) * W + (j0 + jl);
      float vv = alpha * acc[t][r];
      if (beta != 0.f) vv += beta * bf2f(ref[o]);
      Y[o] = f2bf(vv);
    }
  }
}

// ---------------------------------------------------------------------------
// Core of the per-node-weight contraction:
//   acc[n,o] = sum_e emb[n,e] * sum_K xg[b,n,K] * wpool[e,K,o]
// xg rows are split across 3 buffers (cheb orders), each WIDTH wide.
// A tile (16 nodes x 32 K) async-staged to LDS; wpool[e] chunk (32 x DOUT)
// staged K-transposed. 16 nodes / block, DOUT/16 waves. emb scale applied
// per C-row at the end of each e.
// ---------------------------------------------------------------------------
template <int KCH, int DOUT, int WIDTH>
__device__ __forceinline__ void gcn_accum(
    const __bf16* __restrict__ xg0, const __bf16* __restrict__ xg1,
    const __bf16* __restrict__ xg2, const __bf16* __restrict__ wpool,
    const float* __restrict__ emb, int b, int n0,
    __bf16* ldsA, __bf16* ldsB, float* ldsEmb, v8f& accOut) {
  constexpr int CPK = WIDTH / 32;           // 32-chunks per cheb order
  const int tid  = threadIdx.x;
  const int lane = tid & 31, wid = tid >> 5;
  const int jl    = lane & 15;
  const int abase = (lane < 16) ? 0 : 8;
  const int bbase = (lane < 16) ? 0 : 8;
  const int j0    = wid * 16;

  for (int t = tid; t < 16 * EE; t += blockDim.x) {
    const int i = t & 15, e = t >> 4;
    ldsEmb[e * 16 + i] = emb[(n0 + i) * EE + e];
  }
  __syncthreads();

  v8f acc;
#pragma unroll
  for (int r = 0; r < 8; ++r) acc[r] = 0.f;

  for (int e = 0; e < EE; ++e) {
    v8f acce;
#pragma unroll
    for (int r = 0; r < 8; ++r) acce[r] = 0.f;

    for (int kc = 0; kc < KCH; ++kc) {
      const int k   = kc / CPK;
      const int off = (kc - k * CPK) * 32;
      const __bf16* base = (k == 0) ? xg0 : ((k == 1) ? xg1 : xg2);

      // async stage of A (16 rows x 64 bytes = 64 b128 units)
      if (tid < 64) {
        const int mrow = tid >> 2;
        const int part = (tid & 3) * 8;
        async_ld_b128(&ldsA[mrow * 32 + part],
                      base + ((size_t)b * NN + n0 + mrow) * WIDTH + off + part);
      }
      // manual transpose stage of W chunk
      const __bf16* wsrc = wpool + ((size_t)e * KCH + kc) * 32 * DOUT;
      for (int idx = tid; idx < 32 * DOUT; idx += blockDim.x) {
        const int kk = idx / DOUT, j = idx - kk * DOUT;
        ldsB[j * 32 + kk] = wsrc[kk * DOUT + j];
      }
      wait_async0();
      __syncthreads();

      const __bf16* ap = &ldsA[jl * 32 + abase];
      ABfrag af;
      af.h[0] = *(const v8bf*)ap;
      af.h[1] = *(const v8bf*)(ap + 16);

      const __bf16* bp = &ldsB[(j0 + jl) * 32 + bbase];
      ABfrag bf_;
      bf_.h[0] = *(const v8bf*)bp;
      bf_.h[1] = *(const v8bf*)(bp + 16);

      acce = __builtin_amdgcn_wmma_f32_16x16x32_bf16(
          false, af.v, false, bf_.v, (short)0, acce, false, false);
      __syncthreads();
    }

#pragma unroll
    for (int r = 0; r < 8; ++r) {
      const int ml = r + ((lane >= 16) ? 8 : 0);     // C-row within tile
      acc[r] += ldsEmb[e * 16 + ml] * acce[r];
    }
  }
  accOut = acc;
}

// Gate: z_r = sigmoid(gcn + bias). z (o<64) -> cand[...,64+o] bf16; r -> rbuf f32
__global__ void gate_gemm_kernel(const __bf16* __restrict__ xg0,
                                 const __bf16* __restrict__ xg1,
                                 const __bf16* __restrict__ xg2,
                                 const __bf16* __restrict__ wg,
                                 const float* __restrict__ emb,
                                 const float* __restrict__ bias,
                                 __bf16* __restrict__ cand,
                                 float* __restrict__ rbuf) {
  __shared__ __bf16 ldsA[16 * 32];
  __shared__ __bf16 ldsB[32 * 128];
  __shared__ float  ldsEmb[EE * 16];
  const int b = blockIdx.y, n0 = blockIdx.x * 16;
  v8f acc;
  gcn_accum<6, 128, 64>(xg0, xg1, xg2, wg, emb, b, n0, ldsA, ldsB, ldsEmb, acc);

  const int lane = threadIdx.x & 31, wid = threadIdx.x >> 5;
  const int j0 = wid * 16, jl = lane & 15;
#pragma unroll
  for (int r = 0; r < 8; ++r) {
    const int m = n0 + r + ((lane >= 16) ? 8 : 0);
    const int o = j0 + jl;
    float vv = acc[r] + bias[m * 128 + o];
    vv = 1.f / (1.f + __expf(-vv));              // sigmoid
    if (o < 64)
      cand[((size_t)b * NN + m) * 128 + 64 + o] = f2bf(vv);   // z
    else
      rbuf[((size_t)b * NN + m) * 64 + (o - 64)] = vv;        // r
  }
}

// Update: hc = tanh(gcn + bias); h = r*x + (1-r)*hc -> d_out (f32)
__global__ void update_gemm_kernel(const __bf16* __restrict__ xg0,
                                   const __bf16* __restrict__ xg1,
                                   const __bf16* __restrict__ xg2,
                                   const __bf16* __restrict__ wu,
                                   const float* __restrict__ emb,
                                   const float* __restrict__ bias,
                                   const float* __restrict__ rbuf,
                                   const float* __restrict__ xf,
                                   float* __restrict__ out) {
  __shared__ __bf16 ldsA[16 * 32];
  __shared__ __bf16 ldsB[32 * 64];
  __shared__ float  ldsEmb[EE * 16];
  const int b = blockIdx.y, n0 = blockIdx.x * 16;
  v8f acc;
  gcn_accum<12, 64, 128>(xg0, xg1, xg2, wu, emb, b, n0, ldsA, ldsB, ldsEmb, acc);

  const int lane = threadIdx.x & 31, wid = threadIdx.x >> 5;
  const int j0 = wid * 16, jl = lane & 15;
#pragma unroll
  for (int r = 0; r < 8; ++r) {
    const int m = n0 + r + ((lane >= 16) ? 8 : 0);
    const int o = j0 + jl;
    const size_t idx = ((size_t)b * NN + m) * 64 + o;
    const float hc = tanhf(acc[r] + bias[m * 64 + o]);
    const float rv = rbuf[idx];
    out[idx] = rv * xf[idx] + (1.f - rv) * hc;
  }
}

// ---------------------------------------------------------------------------
extern "C" void kernel_launch(void* const* d_in, const int* in_sizes, int n_in,
                              void* d_out, int out_size, void* d_ws,
                              size_t ws_size, hipStream_t stream) {
  (void)in_sizes; (void)n_in; (void)out_size; (void)ws_size;
  const float* x   = (const float*)d_in[0];
  // d_in[1] = state (unused by the reference)
  const float* emb = (const float*)d_in[2];
  const float* gw  = (const float*)d_in[3];
  const float* gb  = (const float*)d_in[4];
  const float* uw  = (const float*)d_in[5];
  const float* ub  = (const float*)d_in[6];
  float* out = (float*)d_out;

  char* w = (char*)d_ws;
  auto alloc = [&](size_t bytes) -> char* {
    char* p = w;
    w += (bytes + 255) & ~(size_t)255;
    return p;
  };
  __bf16* adj   = (__bf16*)alloc((size_t)NN * NN * 2);           //  8 MB
  __bf16* xbf   = (__bf16*)alloc((size_t)BDIM * NN * 64 * 2);    //  8 MB
  __bf16* y1    = (__bf16*)alloc((size_t)BDIM * NN * 64 * 2);    //  8 MB
  __bf16* g2    = (__bf16*)alloc((size_t)BDIM * NN * 64 * 2);    //  8 MB
  __bf16* cand  = (__bf16*)alloc((size_t)BDIM * NN * 128 * 2);   // 16 MB
  __bf16* c1    = (__bf16*)alloc((size_t)BDIM * NN * 128 * 2);   // 16 MB
  __bf16* g2c   = (__bf16*)alloc((size_t)BDIM * NN * 128 * 2);   // 16 MB
  float*  rbuf  = (float*) alloc((size_t)BDIM * NN * 64 * 4);    // 16 MB
  float*  biasg = (float*) alloc((size_t)NN * 128 * 4);
  float*  biasu = (float*) alloc((size_t)NN * 64 * 4);
  __bf16* wgbf  = (__bf16*)alloc((size_t)EE * 3 * 64 * 128 * 2);
  __bf16* wubf  = (__bf16*)alloc((size_t)EE * 3 * 128 * 64 * 2);

  adj_softmax_kernel<<<NN, 256, 0, stream>>>(emb, adj);

  const int nx = BDIM * NN * 64;
  cvt_x_kernel<<<(nx + 255) / 256, 256, 0, stream>>>(x, xbf, cand);
  const int nw = EE * 3 * 64 * 128;
  cvt_f2bf_kernel<<<(nw + 255) / 256, 256, 0, stream>>>(gw, wgbf, nw);
  cvt_f2bf_kernel<<<(nw + 255) / 256, 256, 0, stream>>>(uw, wubf, nw);
  bias_kernel<<<(NN * 128 + 255) / 256, 256, 0, stream>>>(emb, gb, biasg, 128);
  bias_kernel<<<(NN * 64 + 255) / 256, 256, 0, stream>>>(emb, ub, biasu, 64);

  dim3 gs(NN / 32, BDIM);
  // y1 = A@x ; g2 = 2*A@y1 - x
  spmm_kernel<64><<<gs, 256, 0, stream>>>(adj, xbf, 1.f, 0.f, nullptr, y1);
  spmm_kernel<64><<<gs, 256, 0, stream>>>(adj, y1, 2.f, -1.f, xbf, g2);

  dim3 gg(NN / 16, BDIM);
  gate_gemm_kernel<<<gg, 256, 0, stream>>>(xbf, y1, g2, wgbf, emb, biasg,
                                           cand, rbuf);

  // c1 = A@cand ; g2c = 2*A@c1 - cand
  spmm_kernel<128><<<gs, 256, 0, stream>>>(adj, cand, 1.f, 0.f, nullptr, c1);
  spmm_kernel<128><<<gs, 256, 0, stream>>>(adj, c1, 2.f, -1.f, cand, g2c);

  update_gemm_kernel<<<gg, 128, 0, stream>>>(cand, c1, g2c, wubf, emb, biasu,
                                             rbuf, x, out);
}
